// TransformerBlock_19679540150410
// MI455X (gfx1250) — matrix operand order repaired and checked by reading, not compiled, not run on hardware
//
#include <hip/hip_runtime.h>
#include <hip/hip_bf16.h>

// ---------------------------------------------------------------------------
// TransformerBlock on MI455X (gfx1250), wave32 + v_wmma_f32_16x16x32_bf16.
// B=4, N=4096, E=Ei=384.  Flash attention, all GEMMs via WMMA (f32 accum).
//  - GEMM: block stages its 48KB of swizzled weights in LDS once (shared by
//    all 8 waves), 16x64 output per wave, A fragments 1-deep pipelined from
//    global, B fragments from LDS (ds_load_b128, no loop barriers).
//  - Flash: Q fragments register-resident; K^T / transposed-V fragments are
//    contiguous per-lane 32B global loads; P assembled in LDS (cross-wave).
// ---------------------------------------------------------------------------

typedef __bf16 bf16;
typedef __attribute__((ext_vector_type(16))) __bf16 v16bf;
typedef __attribute__((ext_vector_type(8)))  __bf16 v8bf;
typedef __attribute__((ext_vector_type(8)))  float  v8f;

#define EMB   384
#define NSEQ  4096
#define NBAT  4
#define MROWS (NBAT * NSEQ)   // 16384
#define KCH   (EMB / 32)      // 12 k-chunks per 384-deep GEMM

static __device__ __forceinline__ bf16 f2bf(float f) {
  unsigned u = __builtin_bit_cast(unsigned, f);
  unsigned r = (u + 0x7FFFu + ((u >> 16) & 1u)) >> 16;   // RNE
  return __builtin_bit_cast(bf16, (unsigned short)r);
}

static __device__ __forceinline__ v8f wmma_bf16(v16bf a, v16bf b, v8f c) {
  return __builtin_amdgcn_wmma_f32_16x16x32_bf16(
      false, a, false, b, (short)0, c, false, false);
}

// A fragment: 16x32 tile, row-major, lda elements per row.
static __device__ __forceinline__ v16bf load_A_frag(const bf16* __restrict__ p, int lda) {
  const int lane = threadIdx.x & 31;
  const bf16* base = p + (size_t)(lane & 15) * lda + (lane >> 4) * 16;
  v8bf lo = *(const v8bf*)(base);
  v8bf hi = *(const v8bf*)(base + 8);
  return __builtin_shufflevector(lo, hi, 0, 1, 2, 3, 4, 5, 6, 7,
                                         8, 9, 10, 11, 12, 13, 14, 15);
}

// B fragment (32x16, K x N) from an "N-major" source: element (k, n) at
// p[n*stride + k].  One contiguous 32B run per lane.
static __device__ __forceinline__ v16bf load_B_fragN(const bf16* __restrict__ p, int stride) {
  const int lane = threadIdx.x & 31;
  return *(const v16bf*)(p + (size_t)(lane & 15) * stride + (lane >> 4) * 16);
}

// B fragment from fragment-ordered storage: [lane][16] contiguous.
static __device__ __forceinline__ v16bf load_B_fragSW(const bf16* __restrict__ p) {
  const int lane = threadIdx.x & 31;
  return *(const v16bf*)(p + (size_t)lane * 16);
}

// ---------------------------------------------------------------------------
// Weight cast + swizzle into B-fragment order:
//   Wsw[((nt*KCH + kc)*32 + lane)*16 + i] = W[(kc*32 + (lane>>4)*16 + i)*EMB
//                                             + nt*16 + (lane&15)]
// ---------------------------------------------------------------------------
__global__ void __launch_bounds__(256)
swizzle_w_kernel(const float* __restrict__ W, bf16* __restrict__ Wsw) {
  const int o = blockIdx.x * 256 + threadIdx.x;      // 0 .. 147455
  const int i    = o & 15;
  const int lane = (o >> 4) & 31;
  const int rem  = o >> 9;                           // 512-element units
  const int kc   = rem % KCH;
  const int nt   = rem / KCH;
  const int n    = lane & 15;
  const int half = lane >> 4;
  const int k    = kc * 32 + half * 16 + i;
  Wsw[o] = f2bf(W[(size_t)k * EMB + nt * 16 + n]);
}

// ---------------------------------------------------------------------------
// LayerNorm over last dim (384), one block (128 threads) per row, bf16 out.
// ---------------------------------------------------------------------------
__global__ void __launch_bounds__(128)
ln_bf16_kernel(const float* __restrict__ x, const float* __restrict__ gamma,
               const float* __restrict__ beta, bf16* __restrict__ out) {
  const int row = blockIdx.x;
  const int tid = threadIdx.x;
  const float* xr = x + (size_t)row * EMB;
  __shared__ float sred[128];

  float v0 = xr[tid], v1 = xr[tid + 128], v2 = xr[tid + 256];
  sred[tid] = v0 + v1 + v2;
  __syncthreads();
#pragma unroll
  for (int off = 64; off > 0; off >>= 1) {
    if (tid < off) sred[tid] += sred[tid + off];
    __syncthreads();
  }
  const float mu = sred[0] * (1.0f / EMB);
  __syncthreads();

  float d0 = v0 - mu, d1 = v1 - mu, d2 = v2 - mu;
  sred[tid] = d0 * d0 + d1 * d1 + d2 * d2;
  __syncthreads();
#pragma unroll
  for (int off = 64; off > 0; off >>= 1) {
    if (tid < off) sred[tid] += sred[tid + off];
    __syncthreads();
  }
  const float rstd = rsqrtf(sred[0] * (1.0f / EMB) + 1e-5f);

  bf16* o = out + (size_t)row * EMB;
  o[tid]       = f2bf(d0 * rstd * gamma[tid]       + beta[tid]);
  o[tid + 128] = f2bf(d1 * rstd * gamma[tid + 128] + beta[tid + 128]);
  o[tid + 256] = f2bf(d2 * rstd * gamma[tid + 256] + beta[tid + 256]);
}

// ---------------------------------------------------------------------------
// M x 384 x 384 GEMM:  out = A(bf16) @ W(swizzled bf16) + bias [+ resid]
// Block: 8 waves, 48KB of weights staged in LDS once, 16x64 per wave.
// grid = (M/128, 6), block = 256.
// ---------------------------------------------------------------------------
__global__ void __launch_bounds__(256)
gemm384_kernel(const bf16* __restrict__ A, const bf16* __restrict__ Wsw,
               const float* __restrict__ bias, const float* __restrict__ resid,
               float* __restrict__ outf, bf16* __restrict__ outb,
               bf16* __restrict__ outbT) {
  const int lane = threadIdx.x & 31;
  const int wv   = threadIdx.x >> 5;
  const int m0   = blockIdx.x * 128 + wv * 16;
  const int nt0  = blockIdx.y * 4;

  // ---- stage this block's 4 weight n-tiles (48 KB) into LDS -------------
  __shared__ __align__(32) bf16 Wlds[4 * KCH * 512];
  {
    const uint4* src = (const uint4*)(Wsw + (size_t)nt0 * KCH * 512);
    uint4* dst = (uint4*)Wlds;
#pragma unroll
    for (int i = threadIdx.x; i < (4 * KCH * 512) / 8; i += 256)
      dst[i] = src[i];
  }
  __syncthreads();

  const bf16* Arow = A + (size_t)m0 * EMB;

  v8f acc[4] = {{}, {}, {}, {}};
  v16bf a_cur = load_A_frag(Arow, EMB);
#pragma unroll
  for (int kc = 0; kc < KCH; ++kc) {
    v16bf a_nxt = a_cur;
    if (kc + 1 < KCH) a_nxt = load_A_frag(Arow + (kc + 1) * 32, EMB);
#pragma unroll
    for (int t = 0; t < 4; ++t) {
      v16bf b = load_B_fragSW(&Wlds[((size_t)t * KCH + kc) * 512]);
      acc[t] = wmma_bf16(a_cur, b, acc[t]);
    }
    a_cur = a_nxt;
  }

  const int n  = lane & 15;
  const int mb = (lane >> 4) * 8;

#pragma unroll
  for (int t = 0; t < 4; ++t) {
    const int n0 = (nt0 + t) * 16;
    const float bv = bias[n0 + n];
    if (outbT) {
      // transposed store: contiguous along seq per lane
      const int bb  = m0 / NSEQ;
      const int seq = (m0 % NSEQ) + mb;
      bf16* dst = outbT + ((size_t)bb * EMB + n0 + n) * NSEQ + seq;
#pragma unroll
      for (int r = 0; r < 8; ++r) dst[r] = f2bf(acc[t][r] + bv);
    }
    if (outf || outb) {
#pragma unroll
      for (int r = 0; r < 8; ++r) {
        const size_t o = (size_t)(m0 + mb + r) * EMB + n0 + n;
        float v = acc[t][r] + bv;
        if (resid) v += resid[o];
        if (outf)  outf[o] = v;
        if (outb)  outb[o] = f2bf(v);
      }
    }
  }
}

// ---------------------------------------------------------------------------
// Flash attention: S = Q@K^T (no scale, per reference), online softmax,
// ctx = P@V.  Block = 8 waves = one 16-query tile; 128-key blocks.
// grid = (N/16, B)
// ---------------------------------------------------------------------------
__global__ void __launch_bounds__(256)
flash_attn_kernel(const bf16* __restrict__ Q, const bf16* __restrict__ K,
                  const bf16* __restrict__ Vt, bf16* __restrict__ O) {
  const int b    = blockIdx.y;
  const int n0   = blockIdx.x * 16;
  const int tid  = threadIdx.x;
  const int lane = tid & 31;
  const int wv   = tid >> 5;
  const int n    = lane & 15;
  const int mb   = (lane >> 4) * 8;

  __shared__ __align__(32) bf16  Ps[16 * 128];     // 4 KB probability tile
  __shared__ float wred[8][16];
  __shared__ float Mrun[16], Lrun[16], Mnew[16], Cfac[16];

  // preload the 12 Q A-fragments into registers (16 rows x 384)
  const bf16* Qg = Q + ((size_t)b * NSEQ + n0) * EMB;
  v16bf qf[KCH];
#pragma unroll
  for (int kc = 0; kc < KCH; ++kc) qf[kc] = load_A_frag(Qg + kc * 32, EMB);

  if (tid < 16) { Mrun[tid] = -1e30f; Lrun[tid] = 0.0f; }
  __syncthreads();

  v8f acc0 = {}, acc1 = {}, acc2 = {};
  const int c0 = wv * 48;                          // ctx column slice base

  const bf16* Kb  = K  + (size_t)b * NSEQ * EMB;
  const bf16* Vtb = Vt + (size_t)b * EMB * NSEQ;

  for (int j = 0; j < NSEQ; j += 128) {
    // ---- scores: S = Qtile @ K[j+wv*16 .. +16]^T ------------------------
    v8f s = {};
    const bf16* Kt = Kb + (size_t)(j + wv * 16) * EMB;
    v16bf kb_cur = load_B_fragN(Kt, EMB);
#pragma unroll
    for (int kc = 0; kc < KCH; ++kc) {
      v16bf kb_nxt = kb_cur;
      if (kc + 1 < KCH) kb_nxt = load_B_fragN(Kt + (kc + 1) * 32, EMB);
      s = wmma_bf16(qf[kc], kb_cur, s);
      kb_cur = kb_nxt;
    }

    // ---- per-row max across 16 cols (lanes within a half) ---------------
    float mx[8];
#pragma unroll
    for (int r = 0; r < 8; ++r) mx[r] = s[r];
#pragma unroll
    for (int off = 1; off < 16; off <<= 1) {
#pragma unroll
      for (int r = 0; r < 8; ++r)
        mx[r] = fmaxf(mx[r], __shfl_xor(mx[r], off, 32));
    }
    if (n == 0) {
#pragma unroll
      for (int r = 0; r < 8; ++r) wred[wv][mb + r] = mx[r];
    }
    __syncthreads();

    // ---- combine maxes across waves, online-softmax bookkeeping ---------
    if (tid < 16) {
      float bm = wred[0][tid];
#pragma unroll
      for (int w2 = 1; w2 < 8; ++w2) bm = fmaxf(bm, wred[w2][tid]);
      const float mn = fmaxf(Mrun[tid], bm);
      const float cf = __expf(Mrun[tid] - mn);
      Mnew[tid] = mn;
      Cfac[tid] = cf;
      Mrun[tid] = mn;
      Lrun[tid] *= cf;
    }
    __syncthreads();

    // ---- exponentiate, emit bf16 probs to LDS, partial row sums ---------
    float ps[8];
#pragma unroll
    for (int r = 0; r < 8; ++r) {
      const float p = __expf(s[r] - Mnew[mb + r]);
      ps[r] = p;
      Ps[(size_t)(mb + r) * 128 + wv * 16 + n] = f2bf(p);
    }
    float sm[8];
#pragma unroll
    for (int r = 0; r < 8; ++r) sm[r] = ps[r];
#pragma unroll
    for (int off = 1; off < 16; off <<= 1) {
#pragma unroll
      for (int r = 0; r < 8; ++r) sm[r] += __shfl_xor(sm[r], off, 32);
    }
    if (n == 0) {
#pragma unroll
      for (int r = 0; r < 8; ++r) wred[wv][mb + r] = sm[r];
    }
    // rescale running ctx accumulators
#pragma unroll
    for (int r = 0; r < 8; ++r) {
      const float cf = Cfac[mb + r];
      acc0[r] *= cf; acc1[r] *= cf; acc2[r] *= cf;
    }
    __syncthreads();

    if (tid < 16) {
      float ls = 0.0f;
#pragma unroll
      for (int w2 = 0; w2 < 8; ++w2) ls += wred[w2][tid];
      Lrun[tid] += ls;
    }

    // ---- ctx += P(16x128) @ V[j..j+128, c0..c0+48] ----------------------
    const bf16* v0p = Vtb + (size_t)(c0 +  0) * NSEQ + j;
    const bf16* v1p = Vtb + (size_t)(c0 + 16) * NSEQ + j;
    const bf16* v2p = Vtb + (size_t)(c0 + 32) * NSEQ + j;
#pragma unroll
    for (int kc = 0; kc < 4; ++kc) {
      v16bf a = load_A_frag(&Ps[kc * 32], 128);
      acc0 = wmma_bf16(a, load_B_fragN(v0p + kc * 32, NSEQ), acc0);
      acc1 = wmma_bf16(a, load_B_fragN(v1p + kc * 32, NSEQ), acc1);
      acc2 = wmma_bf16(a, load_B_fragN(v2p + kc * 32, NSEQ), acc2);
    }
    __syncthreads();   // Ps / wred / Mnew / Cfac free for next key block
  }

  // ---- normalize by softmax denominator, store bf16 ctx -----------------
  bf16* Og = O + ((size_t)b * NSEQ + n0) * EMB;
#pragma unroll
  for (int r = 0; r < 8; ++r) {
    const int row = mb + r;
    const float inv = 1.0f / Lrun[row];
    Og[(size_t)row * EMB + c0 +  0 + n] = f2bf(acc0[r] * inv);
    Og[(size_t)row * EMB + c0 + 16 + n] = f2bf(acc1[r] * inv);
    Og[(size_t)row * EMB + c0 + 32 + n] = f2bf(acc2[r] * inv);
  }
}

// ---------------------------------------------------------------------------
// Launch
// ---------------------------------------------------------------------------
extern "C" void kernel_launch(void* const* d_in, const int* in_sizes, int n_in,
                              void* d_out, int out_size, void* d_ws, size_t ws_size,
                              hipStream_t stream) {
  const float* x   = (const float*)d_in[0];
  const float* Wq  = (const float*)d_in[1];
  const float* bq  = (const float*)d_in[2];
  const float* Wk  = (const float*)d_in[3];
  const float* bk  = (const float*)d_in[4];
  const float* Wv  = (const float*)d_in[5];
  const float* bv  = (const float*)d_in[6];
  const float* Wa  = (const float*)d_in[7];
  const float* ba  = (const float*)d_in[8];
  const float* Wf  = (const float*)d_in[9];
  const float* bfv = (const float*)d_in[10];
  const float* g1  = (const float*)d_in[11];
  const float* be1 = (const float*)d_in[12];
  const float* g2  = (const float*)d_in[13];
  const float* be2 = (const float*)d_in[14];
  float* out = (float*)d_out;

  char* p = (char*)d_ws;
  auto alloc = [&](size_t bytes) -> char* {
    char* r = p;
    p += (bytes + 255) & ~(size_t)255;
    return r;
  };
  const size_t WEL = (size_t)EMB * EMB;     // 147456
  const size_t ACT = (size_t)MROWS * EMB;   // 6291456

  bf16* Wq_s  = (bf16*)alloc(WEL * 2);
  bf16* Wk_s  = (bf16*)alloc(WEL * 2);
  bf16* Wv_s  = (bf16*)alloc(WEL * 2);
  bf16* Wa_s  = (bf16*)alloc(WEL * 2);
  bf16* Wf_s  = (bf16*)alloc(WEL * 2);
  bf16* h_b   = (bf16*)alloc(ACT * 2);      // LN1 out, reused for LN2 out
  bf16* q_b   = (bf16*)alloc(ACT * 2);
  bf16* k_b   = (bf16*)alloc(ACT * 2);
  bf16* vt_b  = (bf16*)alloc(ACT * 2);      // V transposed [B][E][N]
  bf16* ctx_b = (bf16*)alloc(ACT * 2);
  float* x1   = (float*)alloc(ACT * 4);
  (void)ws_size; (void)in_sizes; (void)n_in; (void)out_size;

  // ---- weights -> bf16, swizzled to fragment order ----
  const int sb = (int)(WEL / 256);
  swizzle_w_kernel<<<sb, 256, 0, stream>>>(Wq, Wq_s);
  swizzle_w_kernel<<<sb, 256, 0, stream>>>(Wk, Wk_s);
  swizzle_w_kernel<<<sb, 256, 0, stream>>>(Wv, Wv_s);
  swizzle_w_kernel<<<sb, 256, 0, stream>>>(Wa, Wa_s);
  swizzle_w_kernel<<<sb, 256, 0, stream>>>(Wf, Wf_s);

  // ---- h = LN1(x) ----
  ln_bf16_kernel<<<MROWS, 128, 0, stream>>>(x, g1, be1, h_b);

  // ---- Q/K/V projections (V written transposed) ----
  dim3 ggrid(MROWS / 128, EMB / 64);
  gemm384_kernel<<<ggrid, 256, 0, stream>>>(h_b, Wq_s, bq, nullptr, nullptr, q_b, nullptr);
  gemm384_kernel<<<ggrid, 256, 0, stream>>>(h_b, Wk_s, bk, nullptr, nullptr, k_b, nullptr);
  gemm384_kernel<<<ggrid, 256, 0, stream>>>(h_b, Wv_s, bv, nullptr, nullptr, nullptr, vt_b);

  // ---- flash attention ----
  flash_attn_kernel<<<dim3(NSEQ / 16, NBAT), 256, 0, stream>>>(q_b, k_b, vt_b, ctx_b);

  // ---- x1 = ctx @ Wa + ba + x ----
  gemm384_kernel<<<ggrid, 256, 0, stream>>>(ctx_b, Wa_s, ba, x, x1, nullptr, nullptr);

  // ---- h2 = LN2(x1) ----
  ln_bf16_kernel<<<MROWS, 128, 0, stream>>>(x1, g2, be2, h_b);

  // ---- out = h2 @ Wf + bf + x1 ----
  gemm384_kernel<<<ggrid, 256, 0, stream>>>(h_b, Wf_s, bfv, x1, out, nullptr, nullptr);
}
